// EnhancTrAISformer_10385230922085
// MI455X (gfx1250) — matrix operand order repaired and checked by reading, not compiled, not run on hardware
//
#include <hip/hip_runtime.h>
#include <hip/hip_bf16.h>
#include <cmath>

typedef __attribute__((ext_vector_type(16))) _Float16 v16h;
typedef __attribute__((ext_vector_type(8)))  _Float16 v8h;
typedef __attribute__((ext_vector_type(8)))  float    v8f;

#define B_  32
#define T_  1024
#define C_  768
#define H_  8
#define D_  96
#define FF_ 3072
#define M_  (B_*T_)          // 32768 rows
#define NLOGIT 622

static __device__ __forceinline__ v16h cat16(v8h lo, v8h hi) {
  return __builtin_shufflevector(lo, hi, 0,1,2,3,4,5,6,7,8,9,10,11,12,13,14,15);
}

// ---------------- f32 [K][N] -> f16 [N][K] transpose-convert ----------------
__global__ __launch_bounds__(256)
void k_convT(const float* __restrict__ W, _Float16* __restrict__ Wt, int K, int N) {
  __shared__ float tile[32][33];
  const int kb = blockIdx.x * 32, nb = blockIdx.y * 32;
  const int tx = threadIdx.x & 31, ty = threadIdx.x >> 5;
  #pragma unroll
  for (int i = ty; i < 32; i += 8) {
    const int k = kb + i, n = nb + tx;
    tile[i][tx] = (k < K && n < N) ? W[(size_t)k * N + n] : 0.f;
  }
  __syncthreads();
  #pragma unroll
  for (int i = ty; i < 32; i += 8) {
    const int n = nb + i, k = kb + tx;
    if (n < N && k < K) Wt[(size_t)n * K + k] = (_Float16)tile[tx][i];
  }
}

// ---------------- concat bq|bk|bv -> [L][2304] ----------------
__global__ __launch_bounds__(256)
void k_bias_concat(const float* __restrict__ bq, const float* __restrict__ bk,
                   const float* __restrict__ bv, float* __restrict__ out) {
  const int l = blockIdx.x, part = blockIdx.y;
  const float* src = (part == 0) ? bq : (part == 1) ? bk : bv;
  for (int i = threadIdx.x; i < C_; i += 256)
    out[(size_t)l * 3 * C_ + (size_t)part * C_ + i] = src[(size_t)l * C_ + i];
}

// ---------------- embedding gather + pos ----------------
__global__ __launch_bounds__(256)
void k_embed(const float* __restrict__ x,
             const float* __restrict__ lat, const float* __restrict__ lon,
             const float* __restrict__ sog, const float* __restrict__ cog,
             const float* __restrict__ pos, float* __restrict__ h) {
  const int bt = blockIdx.x;
  const int t  = bt & (T_ - 1);
  __shared__ int idx[4];
  if (threadIdx.x < 4) {
    const float sz[4] = {250.f, 270.f, 30.f, 72.f};
    idx[threadIdx.x] = (int)(x[(size_t)bt * 4 + threadIdx.x] * sz[threadIdx.x]);
  }
  __syncthreads();
  for (int c = threadIdx.x; c < C_; c += 256) {
    float e;
    if (c < 256)      e = lat[(size_t)idx[0] * 256 + c];
    else if (c < 512) e = lon[(size_t)idx[1] * 256 + (c - 256)];
    else if (c < 640) e = sog[(size_t)idx[2] * 128 + (c - 512)];
    else              e = cog[(size_t)idx[3] * 128 + (c - 640)];
    h[(size_t)bt * C_ + c] = e + pos[(size_t)t * C_ + c];
  }
}

// ---------------- layernorm: one wave per 768-wide row, f16 out ----------------
__global__ __launch_bounds__(256)
void k_layernorm(const float* __restrict__ x, const float* __restrict__ g,
                 const float* __restrict__ bb, _Float16* __restrict__ out) {
  const int row  = blockIdx.x * 8 + (threadIdx.x >> 5);
  const int lane = threadIdx.x & 31;
  const float* xr = x + (size_t)row * C_;
  float v[24], s = 0.f, s2 = 0.f;
  #pragma unroll
  for (int i = 0; i < 24; ++i) { v[i] = xr[lane + i * 32]; s += v[i]; s2 += v[i] * v[i]; }
  #pragma unroll
  for (int off = 1; off < 32; off <<= 1) {
    s  += __shfl_xor(s,  off, 32);
    s2 += __shfl_xor(s2, off, 32);
  }
  const float mean = s * (1.f / C_);
  const float var  = s2 * (1.f / C_) - mean * mean;
  const float rstd = rsqrtf(var + 1e-5f);
  #pragma unroll
  for (int i = 0; i < 24; ++i) {
    const int c = lane + i * 32;
    out[(size_t)row * C_ + c] = (_Float16)((v[i] - mean) * rstd * g[c] + bb[c]);
  }
}

// ---------------- WMMA GEMM: C[M,N] = A[M,K] * Wt[N,K]^T (+bias, epilogue) ----
// Macro tile 64x256, 8 waves as 2(M) x 4(N), each wave 2x4 = 8 WMMAs / k-step.
// Double-buffered LDS (ping-pong): one barrier per k-step, global loads for
// step s+1 issued before the WMMAs of step s.  Compile-time MODE/HAS_BIAS keep
// the epilogue branch-free:
// MODE 0: f16 store   1: f32 +=   2: exact-GELU f16 store   3: f32 store
#define GBM 64
#define GBN 256

template <int MODE, bool HAS_BIAS>
__device__ __forceinline__ void gemm_store(void* out, size_t o, float acc, float bval) {
  float val = acc + (HAS_BIAS ? bval : 0.f);
  if (MODE == 0)      ((_Float16*)out)[o] = (_Float16)val;
  else if (MODE == 1) ((float*)out)[o] += val;
  else if (MODE == 2) ((_Float16*)out)[o] =
      (_Float16)(0.5f * val * (1.f + erff(val * 0.70710678f)));
  else                ((float*)out)[o] = val;
}

template <int MODE, bool HAS_BIAS>
__global__ __launch_bounds__(256)
void k_gemm(const _Float16* __restrict__ A, const _Float16* __restrict__ Wt,
            const float* __restrict__ bias, void* __restrict__ out,
            int N, int K, int ldOut) {
  __shared__ _Float16 As[2][GBM * 40];     // [m][k], stride 40 halfs (80B)
  __shared__ _Float16 Bs[2][GBN * 40];     // [n][k]
  const int tid  = threadIdx.x;
  const int wave = tid >> 5, lane = tid & 31;
  const int wn = wave & 3, wm = wave >> 2;         // 4 x 2 wave grid
  const int hl = lane >> 4, l = lane & 15;
  const int m0 = blockIdx.y * GBM;
  const int n0 = blockIdx.x * GBN;
  const bool fullN = (n0 + GBN) <= N;

  const int ar = tid >> 2, ac = (tid & 3) * 8;     // A: 64 rows x 4 chunks of 8
  const int brb = tid >> 2, bc = (tid & 3) * 8;    // B: rows brb + i*64

  uint4 aR;
  uint4 bR[4];

  // ---- prologue: stage k-step 0 ----
  {
    aR = *(const uint4*)(A + (size_t)(m0 + ar) * K + ac);
    if (fullN) {
      #pragma unroll
      for (int i = 0; i < 4; ++i)
        bR[i] = *(const uint4*)(Wt + (size_t)(n0 + brb + i * 64) * K + bc);
    } else {
      #pragma unroll
      for (int i = 0; i < 4; ++i) {
        const int r = n0 + brb + i * 64;
        bR[i] = (r < N) ? *(const uint4*)(Wt + (size_t)r * K + bc)
                        : make_uint4(0u, 0u, 0u, 0u);
      }
    }
    *(uint4*)(&As[0][ar * 40 + ac]) = aR;
    #pragma unroll
    for (int i = 0; i < 4; ++i)
      *(uint4*)(&Bs[0][(brb + i * 64) * 40 + bc]) = bR[i];
  }

  v8f acc[2][4];
  #pragma unroll
  for (int mt = 0; mt < 2; ++mt)
    #pragma unroll
    for (int nt = 0; nt < 4; ++nt)
      #pragma unroll
      for (int j = 0; j < 8; ++j) acc[mt][nt][j] = 0.f;

  const int steps = K >> 5;
  for (int s = 0; s < steps; ++s) {
    __syncthreads();
    const int p = s & 1;
    const bool more = (s + 1) < steps;
    if (more) {       // issue global loads for next step (overlap with WMMAs)
      const int k1 = (s + 1) << 5;
      aR = *(const uint4*)(A + (size_t)(m0 + ar) * K + k1 + ac);
      if (fullN) {
        #pragma unroll
        for (int i = 0; i < 4; ++i)
          bR[i] = *(const uint4*)(Wt + (size_t)(n0 + brb + i * 64) * K + k1 + bc);
      } else {
        #pragma unroll
        for (int i = 0; i < 4; ++i) {
          const int r = n0 + brb + i * 64;
          bR[i] = (r < N) ? *(const uint4*)(Wt + (size_t)r * K + k1 + bc)
                          : make_uint4(0u, 0u, 0u, 0u);
        }
      }
    }
    // preload ALL fragments, then back-to-back WMMAs (overlapped dscnt waits)
    v16h af[2], bf[4];
    #pragma unroll
    for (int mt = 0; mt < 2; ++mt) {
      const _Float16* ap = &As[p][(wm * 32 + mt * 16 + l) * 40 + hl * 8];
      af[mt] = cat16(*(const v8h*)ap, *(const v8h*)(ap + 16));
    }
    #pragma unroll
    for (int nt = 0; nt < 4; ++nt) {
      const _Float16* bp = &Bs[p][(wn * 64 + nt * 16 + l) * 40 + hl * 8];
      bf[nt] = cat16(*(const v8h*)bp, *(const v8h*)(bp + 16));
    }
    #pragma unroll
    for (int mt = 0; mt < 2; ++mt)
      #pragma unroll
      for (int nt = 0; nt < 4; ++nt)
        acc[mt][nt] = __builtin_amdgcn_wmma_f32_16x16x32_f16(
            false, af[mt], false, bf[nt], (short)0, acc[mt][nt], false, false);
    if (more) {       // stage next step into the other buffer
      const int q = p ^ 1;
      *(uint4*)(&As[q][ar * 40 + ac]) = aR;
      #pragma unroll
      for (int i = 0; i < 4; ++i)
        *(uint4*)(&Bs[q][(brb + i * 64) * 40 + bc]) = bR[i];
    }
  }

  // ---------------- epilogue (branch-free fast path for full tiles) ----------
  if (fullN) {
    #pragma unroll
    for (int mt = 0; mt < 2; ++mt) {
      const int rbase = m0 + wm * 32 + mt * 16 + hl * 8;
      #pragma unroll
      for (int nt = 0; nt < 4; ++nt) {
        const int n = n0 + wn * 64 + nt * 16 + l;
        const float bval = HAS_BIAS ? bias[n] : 0.f;
        #pragma unroll
        for (int vv = 0; vv < 8; ++vv)
          gemm_store<MODE, HAS_BIAS>(out, (size_t)(rbase + vv) * ldOut + n,
                                     acc[mt][nt][vv], bval);
      }
    }
  } else {
    #pragma unroll
    for (int mt = 0; mt < 2; ++mt) {
      const int rbase = m0 + wm * 32 + mt * 16 + hl * 8;
      #pragma unroll
      for (int nt = 0; nt < 4; ++nt) {
        const int n = n0 + wn * 64 + nt * 16 + l;
        if (n >= N) continue;
        const float bval = HAS_BIAS ? bias[n] : 0.f;
        #pragma unroll
        for (int vv = 0; vv < 8; ++vv)
          gemm_store<MODE, HAS_BIAS>(out, (size_t)(rbase + vv) * ldOut + n,
                                     acc[mt][nt][vv], bval);
      }
    }
  }
}

// ---------------- flash attention: 128 q-rows per WG, 32-key blocks ----------
// Per-wave uniform branches skip (a) masking for off-diagonal blocks and
// (b) the entire compute section for fully-masked blocks.  The softmax "l"
// statistic is kept as per-lane partials and reduced once at the end.
__global__ __launch_bounds__(256)
void k_attn(const _Float16* __restrict__ qkv, _Float16* __restrict__ y) {
  const int qb = blockIdx.x;                  // 0..7
  const int b = blockIdx.y >> 3, h = blockIdx.y & 7;
  const int tid = threadIdx.x, wave = tid >> 5, lane = tid & 31;
  const int hl = lane >> 4, l = lane & 15;
  __shared__ _Float16 Kt[32][104];            // [key][d]   (B layout for QK^T)
  __shared__ _Float16 Vt[96][40];             // [d][key]   (B layout for PV)
  __shared__ _Float16 Pl[8][16][32];          // per-wave P scratch

  const size_t rs   = 3 * (size_t)C_;         // 2304
  const size_t base = (size_t)b * T_ * rs + (size_t)h * D_;
  const int qrow0 = qb * 128 + wave * 16;     // wave-uniform

  // Q fragments straight from global (A layout, rows = this wave's 16 q-rows)
  v16h qf[3];
  {
    const _Float16* qp = qkv + base + (size_t)(qrow0 + l) * rs;
    #pragma unroll
    for (int dc = 0; dc < 3; ++dc) {
      const _Float16* p = qp + dc * 32 + hl * 8;
      qf[dc] = cat16(*(const v8h*)p, *(const v8h*)(p + 16));
    }
  }

  v8f o[6];
  float mrow[8], lrow[8];                     // lrow: per-lane partial sums
  #pragma unroll
  for (int dt = 0; dt < 6; ++dt)
    #pragma unroll
    for (int vv = 0; vv < 8; ++vv) o[dt][vv] = 0.f;
  #pragma unroll
  for (int vv = 0; vv < 8; ++vv) { mrow[vv] = -3.0e38f; lrow[vv] = 0.f; }

  const int nkb = (qb + 1) * 4;
  for (int kb = 0; kb < nkb; ++kb) {
    const int ks = kb * 32;
    __syncthreads();
    for (int i = tid; i < 32 * 12; i += 256) {          // K tile
      const int key = i / 12, dc = i % 12;
      *(uint4*)(&Kt[key][dc * 8]) =
          *(const uint4*)(qkv + base + C_ + (size_t)(ks + key) * rs + dc * 8);
    }
    for (int i = tid; i < 32 * 12; i += 256) {          // V tile (transposed)
      const int key = i / 12, dc = i % 12;
      const v8h vv8 = *(const v8h*)(qkv + base + 2 * C_ + (size_t)(ks + key) * rs + dc * 8);
      #pragma unroll
      for (int j = 0; j < 8; ++j) Vt[dc * 8 + j][key] = vv8[j];
    }
    __syncthreads();

    if (ks > qrow0 + 15) continue;            // block fully masked for this wave

    // preload all K and V fragments; V latency hides behind QK^T + softmax
    v16h kf[2][3], vf[6];
    #pragma unroll
    for (int nt = 0; nt < 2; ++nt)
      #pragma unroll
      for (int dc = 0; dc < 3; ++dc) {
        const _Float16* bp = &Kt[nt * 16 + l][dc * 32 + hl * 8];
        kf[nt][dc] = cat16(*(const v8h*)bp, *(const v8h*)(bp + 16));
      }
    #pragma unroll
    for (int dt = 0; dt < 6; ++dt) {
      const _Float16* bp = &Vt[dt * 16 + l][hl * 8];
      vf[dt] = cat16(*(const v8h*)bp, *(const v8h*)(bp + 16));
    }

    v8f s[2];
    #pragma unroll
    for (int nt = 0; nt < 2; ++nt) {
      #pragma unroll
      for (int vv = 0; vv < 8; ++vv) s[nt][vv] = 0.f;
      #pragma unroll
      for (int dc = 0; dc < 3; ++dc)
        s[nt] = __builtin_amdgcn_wmma_f32_16x16x32_f16(false, qf[dc], false, kf[nt][dc],
                                                       (short)0, s[nt], false, false);
    }
    // scale (+ causal mask only on diagonal blocks; wave-uniform branch)
    const float sc = 0.1020620726f;           // 1/sqrt(96)
    if (ks + 31 > qrow0) {
      #pragma unroll
      for (int nt = 0; nt < 2; ++nt)
        #pragma unroll
        for (int vv = 0; vv < 8; ++vv) {
          const int kidx = ks + nt * 16 + l;
          const int qidx = qrow0 + hl * 8 + vv;
          float val = s[nt][vv] * sc;
          if (kidx > qidx) val = -3.0e38f;
          s[nt][vv] = val;
        }
    } else {
      #pragma unroll
      for (int nt = 0; nt < 2; ++nt)
        #pragma unroll
        for (int vv = 0; vv < 8; ++vv) s[nt][vv] *= sc;
    }
    // online softmax: only the row-max needs a butterfly; "l" stays per-lane
    float corr[8];
    #pragma unroll
    for (int vv = 0; vv < 8; ++vv) {
      float rmax = fmaxf(s[0][vv], s[1][vv]);
      #pragma unroll
      for (int off = 1; off < 16; off <<= 1) rmax = fmaxf(rmax, __shfl_xor(rmax, off, 32));
      const float mnew = fmaxf(mrow[vv], rmax);
      corr[vv] = __expf(mrow[vv] - mnew);
      mrow[vv] = mnew;
      const float p0 = __expf(s[0][vv] - mnew);
      const float p1 = __expf(s[1][vv] - mnew);
      s[0][vv] = p0; s[1][vv] = p1;
      lrow[vv] = lrow[vv] * corr[vv] + (p0 + p1);
    }
    // C-layout P -> A-layout via per-wave LDS
    #pragma unroll
    for (int nt = 0; nt < 2; ++nt)
      #pragma unroll
      for (int vv = 0; vv < 8; ++vv)
        Pl[wave][hl * 8 + vv][nt * 16 + l] = (_Float16)s[nt][vv];
    asm volatile("s_wait_dscnt 0" ::: "memory");        // same-wave LDS RAW
    const _Float16* pp = &Pl[wave][l][hl * 8];
    const v16h pf = cat16(*(const v8h*)pp, *(const v8h*)(pp + 16));
    #pragma unroll
    for (int dt = 0; dt < 6; ++dt) {
      #pragma unroll
      for (int vv = 0; vv < 8; ++vv) o[dt][vv] *= corr[vv];
      o[dt] = __builtin_amdgcn_wmma_f32_16x16x32_f16(false, pf, false, vf[dt],
                                                     (short)0, o[dt], false, false);
    }
  }
  // final cross-lane reduction of l, then normalize + store
  float inv[8];
  #pragma unroll
  for (int vv = 0; vv < 8; ++vv) {
    float lsum = lrow[vv];
    #pragma unroll
    for (int off = 1; off < 16; off <<= 1) lsum += __shfl_xor(lsum, off, 32);
    inv[vv] = 1.0f / lsum;
  }
  #pragma unroll
  for (int dt = 0; dt < 6; ++dt)
    #pragma unroll
    for (int vv = 0; vv < 8; ++vv) {
      const size_t row = (size_t)b * T_ + qrow0 + hl * 8 + vv;
      y[row * C_ + h * D_ + dt * 16 + l] = (_Float16)(o[dt][vv] * inv[vv]);
    }
}

// =============================== host launcher ===============================
extern "C" void kernel_launch(void* const* d_in, const int* in_sizes, int n_in,
                              void* d_out, int out_size, void* d_ws, size_t ws_size,
                              hipStream_t stream) {
  (void)in_sizes; (void)n_in; (void)out_size; (void)ws_size;
  const float* x     = (const float*)d_in[0];
  const float* lat   = (const float*)d_in[1];
  const float* lon   = (const float*)d_in[2];
  const float* sog   = (const float*)d_in[3];
  const float* cog   = (const float*)d_in[4];
  const float* pos   = (const float*)d_in[5];
  const float* ln1g  = (const float*)d_in[6];
  const float* ln1b  = (const float*)d_in[7];
  const float* Wq    = (const float*)d_in[8];
  const float* bq    = (const float*)d_in[9];
  const float* Wk    = (const float*)d_in[10];
  const float* bk    = (const float*)d_in[11];
  const float* Wv    = (const float*)d_in[12];
  const float* bv    = (const float*)d_in[13];
  const float* Wo    = (const float*)d_in[14];
  const float* bo    = (const float*)d_in[15];
  const float* ln2g  = (const float*)d_in[16];
  const float* ln2b  = (const float*)d_in[17];
  const float* W1    = (const float*)d_in[18];
  const float* b1    = (const float*)d_in[19];
  const float* W2    = (const float*)d_in[20];
  const float* b2    = (const float*)d_in[21];
  const float* lnfg  = (const float*)d_in[22];
  const float* lnfb  = (const float*)d_in[23];
  const float* Whead = (const float*)d_in[24];

  char* w = (char*)d_ws;
  auto alloc = [&](size_t bytes) -> char* {
    char* p = w; w += (bytes + 255) & ~(size_t)255; return p;
  };
  float*    hbuf   = (float*)   alloc((size_t)M_ * C_ * 4);
  _Float16* xn     = (_Float16*)alloc((size_t)M_ * C_ * 2);
  _Float16* qkv    = (_Float16*)alloc((size_t)M_ * 3 * C_ * 2);
  _Float16* ybuf   = (_Float16*)alloc((size_t)M_ * C_ * 2);
  _Float16* ffbuf  = (_Float16*)alloc((size_t)M_ * FF_ * 2);
  _Float16* Wqkv_t = (_Float16*)alloc((size_t)8 * 3 * C_ * C_ * 2);
  _Float16* Wo_t   = (_Float16*)alloc((size_t)8 * C_ * C_ * 2);
  _Float16* W1_t   = (_Float16*)alloc((size_t)8 * FF_ * C_ * 2);
  _Float16* W2_t   = (_Float16*)alloc((size_t)8 * C_ * FF_ * 2);
  _Float16* Wh_t   = (_Float16*)alloc((size_t)NLOGIT * C_ * 2);
  float*    bqkv   = (float*)   alloc((size_t)8 * 3 * C_ * 4);

  const dim3 cb(256);
  for (int lyr = 0; lyr < 8; ++lyr) {
    const size_t wsz = (size_t)C_ * C_;
    _Float16* dst = Wqkv_t + (size_t)lyr * 3 * C_ * C_;
    k_convT<<<dim3(C_/32, C_/32),  cb, 0, stream>>>(Wq + lyr * wsz, dst,                C_, C_);
    k_convT<<<dim3(C_/32, C_/32),  cb, 0, stream>>>(Wk + lyr * wsz, dst + wsz,          C_, C_);
    k_convT<<<dim3(C_/32, C_/32),  cb, 0, stream>>>(Wv + lyr * wsz, dst + 2 * wsz,      C_, C_);
    k_convT<<<dim3(C_/32, C_/32),  cb, 0, stream>>>(Wo + lyr * wsz, Wo_t + lyr * wsz,   C_, C_);
    k_convT<<<dim3(C_/32, FF_/32), cb, 0, stream>>>(W1 + (size_t)lyr * C_ * FF_,
                                                    W1_t + (size_t)lyr * FF_ * C_,      C_, FF_);
    k_convT<<<dim3(FF_/32, C_/32), cb, 0, stream>>>(W2 + (size_t)lyr * FF_ * C_,
                                                    W2_t + (size_t)lyr * C_ * FF_,      FF_, C_);
  }
  k_convT<<<dim3(C_/32, (NLOGIT + 31) / 32), cb, 0, stream>>>(Whead, Wh_t, C_, NLOGIT);
  k_bias_concat<<<dim3(8, 3), cb, 0, stream>>>(bq, bk, bv, bqkv);

  k_embed<<<dim3(M_), cb, 0, stream>>>(x, lat, lon, sog, cog, pos, hbuf);

  for (int lyr = 0; lyr < 8; ++lyr) {
    k_layernorm<<<dim3(M_/8), cb, 0, stream>>>(hbuf, ln1g + lyr * C_, ln1b + lyr * C_, xn);
    k_gemm<0, true><<<dim3(3*C_/GBN, M_/GBM), cb, 0, stream>>>(
        xn, Wqkv_t + (size_t)lyr * 3 * C_ * C_, bqkv + (size_t)lyr * 3 * C_,
        qkv, 3 * C_, C_, 3 * C_);
    k_attn<<<dim3(T_/128, B_*H_), cb, 0, stream>>>(qkv, ybuf);
    k_gemm<1, true><<<dim3(C_/GBN, M_/GBM), cb, 0, stream>>>(
        ybuf, Wo_t + (size_t)lyr * C_ * C_, bo + lyr * C_, hbuf, C_, C_, C_);
    k_layernorm<<<dim3(M_/8), cb, 0, stream>>>(hbuf, ln2g + lyr * C_, ln2b + lyr * C_, xn);
    k_gemm<2, true><<<dim3(FF_/GBN, M_/GBM), cb, 0, stream>>>(
        xn, W1_t + (size_t)lyr * FF_ * C_, b1 + lyr * FF_, ffbuf, FF_, C_, FF_);
    k_gemm<1, true><<<dim3(C_/GBN, M_/GBM), cb, 0, stream>>>(
        ffbuf, W2_t + (size_t)lyr * C_ * FF_, b2 + lyr * C_, hbuf, C_, FF_, C_);
  }
  k_layernorm<<<dim3(M_/8), cb, 0, stream>>>(hbuf, lnfg, lnfb, xn);
  k_gemm<3, false><<<dim3((NLOGIT + GBN - 1) / GBN, M_/GBM), cb, 0, stream>>>(
      xn, Wh_t, nullptr, d_out, NLOGIT, C_, NLOGIT);
}
